// LSTMD_8280696947210
// MI455X (gfx1250) — compile-verified, matrix operand order checked by reading, not run on hardware
//
#include <hip/hip_runtime.h>

// ---------------------------------------------------------------------------
// LSTM-D recurrence for MI455X (gfx1250, wave32, WMMA bf16).
// B=128, T=256, S=H=320, IN=960.
// 8 persistent workgroups (one per 16-row batch tile) x 640 threads (20 waves,
// one wave per 16-col N-tile). h/C state lives in registers; xhm lives in LDS
// as bf16; weights are pre-packed to WMMA B-fragment layout (4 gates
// interleaved per k-stage) and streamed from L2 through a register pipeline.
// ---------------------------------------------------------------------------

#define B_   128
#define T_   256
#define S_   320
#define H_   320
#define IN_  960
#define NT   20     // H_/16 output column tiles
#define KT_G 30     // IN_/32 k-steps for gate GEMMs
#define KT_H 10     // S_/32  k-steps for decay GEMM
#define XHM_STRIDE 968   // 960 + 8 bf16 pad: 484 dwords/row -> conflict-free
#define D_STRIDE   328   // 320 + 8 bf16 pad

typedef __attribute__((ext_vector_type(16))) __bf16 v16bf;
typedef __attribute__((ext_vector_type(8)))  float  v8f;

union Frag16 { v16bf v; unsigned int w[8]; unsigned short u[16]; };

#define WMMA_BF16(A, Bf, C) \
    __builtin_amdgcn_wmma_f32_16x16x32_bf16(false, (A).v, false, (Bf).v, (short)0, (C), false, false)

__device__ __forceinline__ unsigned short f2bf(float f) {
    unsigned int x = __float_as_uint(f);
    x += 0x7fffu + ((x >> 16) & 1u);          // round-to-nearest-even
    return (unsigned short)(x >> 16);
}
__device__ __forceinline__ float sigmoidf_(float x) {
    return 1.0f / (1.0f + __expf(-x));
}
__device__ __forceinline__ float tanh_fast(float x) {
    x = fminf(fmaxf(x, -15.0f), 15.0f);       // avoid inf/inf
    float e = __expf(2.0f * x);
    return (e - 1.0f) / (e + 1.0f);
}
// 16-bit A/B fragment in-lane k index (ISA 05_wmma 16-bit 16x32 layout):
// lanes 0-15 : vgpr p<4 -> K = 2p+e, p>=4 -> K = 16 + 2(p-4)+e
// lanes 16-31: same + 8
__device__ __forceinline__ int k_of(int hi, int j) {
    int p = j >> 1, e = j & 1;
    return (p < 4) ? (hi * 8 + p * 2 + e) : (16 + hi * 8 + (p - 4) * 2 + e);
}

// A fragment: 8 dword gathers from an LDS bf16 row (row ptr already at k base)
__device__ __forceinline__ void load_afrag(Frag16& A, const unsigned short* row, int hi) {
    #pragma unroll
    for (int p = 0; p < 8; ++p) {
        int kp = (p < 4) ? (hi * 8 + p * 2) : (16 + hi * 8 + (p - 4) * 2);
        A.w[p] = *(const unsigned int*)&row[kp];
    }
}
// B fragment: 32 contiguous bytes per lane (fb already includes lane*16)
__device__ __forceinline__ void load_bfrag(Frag16& Bf, const unsigned short* fb) {
    *(uint4*)&Bf.w[0] = *(const uint4*)(fb);
    *(uint4*)&Bf.w[4] = *(const uint4*)(fb + 8);
}
__device__ __forceinline__ void load_b4(Frag16* Bs, const unsigned short* sb) {
    #pragma unroll
    for (int g = 0; g < 4; ++g) load_bfrag(Bs[g], sb + g * 512);
}

// Pack W_gh (rows=N, cols=K) into bf16 B-fragments, stage-major.
__global__ void pack_b_kernel(const float* __restrict__ W, unsigned short* __restrict__ Wp,
                              int ktiles, int K, int total) {
    int idx = blockIdx.x * blockDim.x + threadIdx.x;
    if (idx >= total) return;
    int j    = idx & 15;
    int lane = (idx >> 4) & 31;
    int frag = idx >> 9;
    int kt   = frag % ktiles;
    int nt   = frag / ktiles;
    int n = nt * 16 + (lane & 15);
    int k = kt * 32 + k_of(lane >> 4, j);
    Wp[idx] = f2bf(W[(size_t)n * K + k]);
}

// Pack the 4 gate weight matrices interleaved: one (nt,kt) "stage" is 4KB =
// 4 gates x 512-bf16 fragments, so the hot loop streams contiguous stages.
// idx = ((nt*KT_G + kt)*4 + g)*512 + lane*16 + j
__global__ void pack_gate4_kernel(const float* __restrict__ Wf, const float* __restrict__ Wi,
                                  const float* __restrict__ Wo, const float* __restrict__ Wc,
                                  unsigned short* __restrict__ Wp, int total) {
    int idx = blockIdx.x * blockDim.x + threadIdx.x;
    if (idx >= total) return;
    int j    = idx & 15;
    int lane = (idx >> 4) & 31;
    int g    = (idx >> 9) & 3;
    int frag = idx >> 11;
    int kt   = frag % KT_G;
    int nt   = frag / KT_G;
    const float* W = (g == 0) ? Wf : (g == 1) ? Wi : (g == 2) ? Wo : Wc;
    int n = nt * 16 + (lane & 15);
    int k = kt * 32 + k_of(lane >> 4, j);
    Wp[idx] = f2bf(W[(size_t)n * IN_ + k]);
}

__global__ __launch_bounds__(640) void lstmd_kernel(
    const float* __restrict__ inp,   const float* __restrict__ Xmean,
    const float* __restrict__ b_f,   const float* __restrict__ b_i,
    const float* __restrict__ b_o,   const float* __restrict__ b_c,
    const float* __restrict__ W_gx,  const float* __restrict__ b_gx,
    const float* __restrict__ b_gh,
    const unsigned short* __restrict__ Wg4,   // interleaved 4-gate fragments
    const unsigned short* __restrict__ Wpgh,  // decay GEMM fragments
    float* __restrict__ out)
{
    __shared__ unsigned short xhm[16 * XHM_STRIDE];  // bf16 [x | h | m]
    __shared__ unsigned short dls[16 * D_STRIDE];    // bf16 delta tile
    __shared__ float sb_gx[S_], s_wgxd[S_], sb_gh[S_];
    __shared__ float sb_f[S_], sb_i[S_], sb_o[S_], sb_c[S_];

    const int tid  = threadIdx.x;
    const int b0   = blockIdx.x * 16;
    const int lane = tid & 31;
    const int w    = tid >> 5;       // wave id == N-tile id (0..19)
    const int nl   = lane & 15;
    const int hi   = lane >> 4;
    const int n    = w * 16 + nl;    // this lane's output column

    if (tid < S_) {
        sb_gx[tid]  = b_gx[tid];
        s_wgxd[tid] = W_gx[(size_t)tid * S_ + tid];  // eye(S)*W_gx -> diagonal
        sb_gh[tid]  = b_gh[tid];
        sb_f[tid] = b_f[tid]; sb_i[tid] = b_i[tid];
        sb_o[tid] = b_o[tid]; sb_c[tid] = b_c[tid];
    }

    float hreg[8], Creg[8];                 // rows m = r + hi*8, col n
    #pragma unroll
    for (int r = 0; r < 8; ++r) { hreg[r] = 0.0f; Creg[r] = 0.0f; }
    __syncthreads();

    const size_t TS = (size_t)T_ * S_;
    const unsigned short* wb_base = Wg4  + (size_t)w * (KT_G * 2048) + lane * 16;
    const unsigned short* db_base = Wpgh + (size_t)w * (KT_H * 512)  + lane * 16;

    for (int t = 0; t < T_; ++t) {
        // ---- Phase A: decay-x imputation, fill xhm[x], xhm[m], d tile -----
        #pragma unroll
        for (int j = 0; j < 8; ++j) {
            int e = tid + j * 640;          // 5120 = 16*320 elements
            int m = e / S_, s = e - m * S_;
            size_t base = (size_t)(b0 + m) * 4 * TS + (size_t)t * S_ + s;
            float x   = inp[base];
            float msk = inp[base + TS];
            float d   = inp[base + 2 * TS];
            float xl  = inp[base + 3 * TS];
            float xm  = Xmean[t * S_ + s];
            if (lane == 0 && t + 1 < T_) {  // next-step lines (t stride = S_)
                __builtin_prefetch(&inp[base + S_], 0, 3);
                __builtin_prefetch(&inp[base + TS + S_], 0, 3);
                __builtin_prefetch(&inp[base + 2 * TS + S_], 0, 3);
                __builtin_prefetch(&inp[base + 3 * TS + S_], 0, 3);
            }
            float dx = __expf(-fmaxf(0.0f, d * s_wgxd[s] + sb_gx[s]));
            float xe = msk * x + (1.0f - msk) * (dx * xl + (1.0f - dx) * xm);
            xhm[m * XHM_STRIDE + s]            = f2bf(xe);
            xhm[m * XHM_STRIDE + 2 * S_ + s]   = f2bf(msk);
            dls[m * D_STRIDE + s]              = f2bf(d);
        }
        __syncthreads();

        // ---- Phase B: delta_h = exp(-relu(d @ Wgh^T + bgh)); h *= delta_h --
        // Register pipeline over the 10 k-stages. A scalar zero offset is
        // laundered through volatile asm each timestep: keeps pointer
        // provenance (loads stay GLOBAL, saddr form) while preventing LICM
        // from hoisting the t-invariant fragment loads out of the t-loop.
        {
            size_t zoff = 0;
            asm volatile("" : "+s"(zoff));
            const unsigned short* db  = db_base + zoff;
            const unsigned short* dr0 = &dls[nl * D_STRIDE];
            v8f acc = {};
            Frag16 D0, D1, D2, D3, a0, a1, a2, a3;
            load_bfrag(D0, db);            load_afrag(a0, dr0,       hi);
            load_bfrag(D1, db + 512);      load_afrag(a1, dr0 + 32,  hi);
            load_bfrag(D2, db + 1024);     load_afrag(a2, dr0 + 64,  hi);
            load_bfrag(D3, db + 1536);     load_afrag(a3, dr0 + 96,  hi);
            // consume 0-3, refill 4-7
            acc = WMMA_BF16(a0, D0, acc); load_bfrag(D0, db + 4*512); load_afrag(a0, dr0 + 4*32, hi);
            acc = WMMA_BF16(a1, D1, acc); load_bfrag(D1, db + 5*512); load_afrag(a1, dr0 + 5*32, hi);
            acc = WMMA_BF16(a2, D2, acc); load_bfrag(D2, db + 6*512); load_afrag(a2, dr0 + 6*32, hi);
            acc = WMMA_BF16(a3, D3, acc); load_bfrag(D3, db + 7*512); load_afrag(a3, dr0 + 7*32, hi);
            // consume 4-7, refill 8-9
            acc = WMMA_BF16(a0, D0, acc); load_bfrag(D0, db + 8*512); load_afrag(a0, dr0 + 8*32, hi);
            acc = WMMA_BF16(a1, D1, acc); load_bfrag(D1, db + 9*512); load_afrag(a1, dr0 + 9*32, hi);
            acc = WMMA_BF16(a2, D2, acc);
            acc = WMMA_BF16(a3, D3, acc);
            // consume 8-9
            acc = WMMA_BF16(a0, D0, acc);
            acc = WMMA_BF16(a1, D1, acc);

            #pragma unroll
            for (int r = 0; r < 8; ++r) {
                int m = r + hi * 8;
                float dh = __expf(-fmaxf(0.0f, acc[r] + sb_gh[n]));
                float hn = dh * hreg[r];
                hreg[r] = hn;
                xhm[m * XHM_STRIDE + S_ + n] = f2bf(hn);   // xhm[h] slot
            }
        }
        __syncthreads();

        // ---- Phase C: 4 gate GEMMs + cell update (all local to this wave) --
        // 3-slot rotating register pipeline; each slot = 4 gate fragments
        // (one 4KB contiguous stage). Loads run 3 stages ahead of the wmmas.
        {
            size_t zoff = 0;
            asm volatile("" : "+s"(zoff));
            const unsigned short* wb  = wb_base + zoff;
            const unsigned short* xr0 = &xhm[nl * XHM_STRIDE];
            v8f af = {}, ai = {}, ao = {}, ac = {};
            Frag16 B0[4], B1[4], B2[4], A0, A1, A2;
            load_b4(B0, wb);          load_afrag(A0, xr0,      hi);
            load_b4(B1, wb + 2048);   load_afrag(A1, xr0 + 32, hi);
            load_b4(B2, wb + 4096);   load_afrag(A2, xr0 + 64, hi);
            #pragma clang loop unroll(disable)
            for (int mk = 0; mk < 9; ++mk) {
                int s = mk * 3;
                af = WMMA_BF16(A0, B0[0], af); ai = WMMA_BF16(A0, B0[1], ai);
                ao = WMMA_BF16(A0, B0[2], ao); ac = WMMA_BF16(A0, B0[3], ac);
                load_b4(B0, wb + (size_t)(s + 3) * 2048);
                load_afrag(A0, xr0 + (s + 3) * 32, hi);
                af = WMMA_BF16(A1, B1[0], af); ai = WMMA_BF16(A1, B1[1], ai);
                ao = WMMA_BF16(A1, B1[2], ao); ac = WMMA_BF16(A1, B1[3], ac);
                load_b4(B1, wb + (size_t)(s + 4) * 2048);
                load_afrag(A1, xr0 + (s + 4) * 32, hi);
                af = WMMA_BF16(A2, B2[0], af); ai = WMMA_BF16(A2, B2[1], ai);
                ao = WMMA_BF16(A2, B2[2], ao); ac = WMMA_BF16(A2, B2[3], ac);
                load_b4(B2, wb + (size_t)(s + 5) * 2048);
                load_afrag(A2, xr0 + (s + 5) * 32, hi);
            }
            // tail: stages 27, 28, 29 already resident
            af = WMMA_BF16(A0, B0[0], af); ai = WMMA_BF16(A0, B0[1], ai);
            ao = WMMA_BF16(A0, B0[2], ao); ac = WMMA_BF16(A0, B0[3], ac);
            af = WMMA_BF16(A1, B1[0], af); ai = WMMA_BF16(A1, B1[1], ai);
            ao = WMMA_BF16(A1, B1[2], ao); ac = WMMA_BF16(A1, B1[3], ac);
            af = WMMA_BF16(A2, B2[0], af); ai = WMMA_BF16(A2, B2[1], ai);
            ao = WMMA_BF16(A2, B2[2], ao); ac = WMMA_BF16(A2, B2[3], ac);

            #pragma unroll
            for (int r = 0; r < 8; ++r) {
                float fv = sigmoidf_(af[r] + sb_f[n]);
                float iv = sigmoidf_(ai[r] + sb_i[n]);
                float ov = sigmoidf_(ao[r] + sb_o[n]);
                float cv = tanh_fast(ac[r] + sb_c[n]);
                float Cn = fv * Creg[r] + iv * cv;
                Creg[r] = Cn;
                hreg[r] = ov * tanh_fast(Cn);
            }
        }
        __syncthreads();
    }

    // final h -> out (B, H) fp32; each lane owns 8 (m, n) cells
    #pragma unroll
    for (int r = 0; r < 8; ++r) {
        int m = r + hi * 8;
        out[(size_t)(b0 + m) * H_ + n] = hreg[r];
    }
}

extern "C" void kernel_launch(void* const* d_in, const int* in_sizes, int n_in,
                              void* d_out, int out_size, void* d_ws, size_t ws_size,
                              hipStream_t stream)
{
    const float* inp   = (const float*)d_in[0];
    const float* Xmean = (const float*)d_in[1];
    const float* W_f   = (const float*)d_in[2];  const float* b_f  = (const float*)d_in[3];
    const float* W_i   = (const float*)d_in[4];  const float* b_i  = (const float*)d_in[5];
    const float* W_o   = (const float*)d_in[6];  const float* b_o  = (const float*)d_in[7];
    const float* W_c   = (const float*)d_in[8];  const float* b_c  = (const float*)d_in[9];
    const float* W_gx  = (const float*)d_in[10]; const float* b_gx = (const float*)d_in[11];
    const float* W_gh  = (const float*)d_in[12]; const float* b_gh = (const float*)d_in[13];

    // Workspace: interleaved 4-gate weights (4 x 320x960) + W_gh (320x320)
    // as packed bf16 fragments = 2,662,400 bytes total.
    unsigned short* ws = (unsigned short*)d_ws;
    unsigned short* Wg4  = ws;                                  // 1,228,800 elems
    unsigned short* Wpgh = Wg4 + (size_t)4 * H_ * IN_;          //   102,400 elems

    const int totG4 = NT * KT_G * 4 * 512;               // 1228800
    const int totH  = NT * KT_H * 512;                   //  102400
    pack_gate4_kernel<<<(totG4 + 255) / 256, 256, 0, stream>>>(W_f, W_i, W_o, W_c, Wg4, totG4);
    pack_b_kernel<<<(totH + 255) / 256, 256, 0, stream>>>(W_gh, Wpgh, KT_H, S_, totH);

    lstmd_kernel<<<B_ / 16, 640, 0, stream>>>(
        inp, Xmean, b_f, b_i, b_o, b_c, W_gx, b_gx, b_gh,
        Wg4, Wpgh, (float*)d_out);
}
